// TemporalSelfAttention_13932873909055
// MI455X (gfx1250) — matrix operand order repaired and checked by reading, not compiled, not run on hardware
//
#include <hip/hip_runtime.h>
#include <hip/hip_bf16.h>

typedef __attribute__((ext_vector_type(16))) _Float16 v16h;
typedef __attribute__((ext_vector_type(8)))  _Float16 v8h;
typedef __attribute__((ext_vector_type(8)))  float    v8f;

#define NQ   16384
#define EMB  256
#define HH   8
#define DH   32
#define PP   4
#define NB   2
#define GW   128
#define GH   128
#define QE_K 512

// ---------------------------------------------------------------------------
// elementwise prep kernels
// ---------------------------------------------------------------------------
__global__ void k_f32_to_f16(const float* __restrict__ src,
                             _Float16* __restrict__ dst, int n) {
  int i = blockIdx.x * blockDim.x + threadIdx.x;
  if (i < n) dst[i] = (_Float16)src[i];
}

// qe[q, 0:256] = value[q, :], qe[q, 256:512] = query[q, :]  (f16)
__global__ void k_build_qe(const float* __restrict__ value,
                           const float* __restrict__ query,
                           _Float16* __restrict__ qe) {
  int i = blockIdx.x * blockDim.x + threadIdx.x;   // over NQ*512
  if (i >= NQ * QE_K) return;
  int q = i >> 9, c = i & 511;
  float x = (c < EMB) ? value[q * EMB + c] : query[q * EMB + (c - EMB)];
  qe[i] = (_Float16)x;
}

// Wt[n*K + k] = (f16) W[k*N + n]   (W is [K x N] row-major)
__global__ void k_transpose_f16(const float* __restrict__ W,
                                _Float16* __restrict__ Wt, int K, int N) {
  int i = blockIdx.x * blockDim.x + threadIdx.x;   // i = n*K + k
  if (i >= N * K) return;
  int n = i / K, k = i - n * K;
  Wt[i] = (_Float16)W[k * N + n];
}

// ---------------------------------------------------------------------------
// generic WMMA f16->f32 GEMM:  out = A[MxK] * Bt[NxK]^T + bias (+ resid)
// one wave computes a 16x64 strip (4 N-tiles). M%16==0, N%64==0, K%32==0,
// K>=64. Software-pipelined: next K-chunk's fragments are loaded while the
// current chunk's 4 WMMAs execute.
// ---------------------------------------------------------------------------
__global__ void __launch_bounds__(256)
k_gemm_wmma(const _Float16* __restrict__ A,
            const _Float16* __restrict__ Bt,
            const float* __restrict__ bias,
            const float* __restrict__ resid,
            float* __restrict__ outF,
            _Float16* __restrict__ outH,
            int M, int N, int K) {
  const int lane = threadIdx.x & 31;
  const int wave = (int)((blockIdx.x * blockDim.x + threadIdx.x) >> 5);
  const int nN64 = N >> 6;
  const int tm = wave / nN64;
  const int tn = wave % nN64;
  if (tm >= (M >> 4)) return;              // wave-uniform: EXEC stays all-ones

  const int l15  = lane & 15;
  const int hiH  = (lane & 16) ? 1 : 0;
  const int rowA = (tm << 4) + l15;
  const int kbA  = hiH ? 8 : 0;            // A-frag K base within 32-chunk
  const int kbB  = hiH ? 8 : 0;            // B-frag K base within 32-chunk
  const int colBase = tn << 6;

  const _Float16* Arow = A + (size_t)rowA * K + kbA;
  const _Float16* Bcol[4];
  #pragma unroll
  for (int j = 0; j < 4; ++j)
    Bcol[j] = Bt + (size_t)(colBase + (j << 4) + l15) * K + kbB;

  v8f acc[4];
  acc[0] = (v8f){}; acc[1] = (v8f){}; acc[2] = (v8f){}; acc[3] = (v8f){};

  // fragment loaders for K-chunk base k0
  auto loadA = [&](int k0) -> v16h {
    const v8h lo = *(const v8h*)(Arow + k0);        // K = kbA+0..7
    const v8h hi = *(const v8h*)(Arow + k0 + 16);   // K = kbA+16..23
    v16h a;
    #pragma unroll
    for (int i = 0; i < 8; ++i) { a[i] = lo[i]; a[8 + i] = hi[i]; }
    return a;
  };
  auto loadB = [&](int j, int k0) -> v16h {
    const v8h lo = *(const v8h*)(Bcol[j] + k0);     // K = kbB+0..7
    const v8h hi = *(const v8h*)(Bcol[j] + k0 + 16); // K = kbB+16..23
    v16h b;
    #pragma unroll
    for (int i = 0; i < 8; ++i) { b[i] = lo[i]; b[8 + i] = hi[i]; }
    return b;
  };

  // prologue
  v16h a = loadA(0);
  v16h b0 = loadB(0, 0), b1 = loadB(1, 0), b2 = loadB(2, 0), b3 = loadB(3, 0);

  // pipelined main loop: issue loads for chunk k0, then WMMA the previous one
  for (int k0 = 32; k0 < K; k0 += 32) {
    v16h an  = loadA(k0);
    v16h bn0 = loadB(0, k0), bn1 = loadB(1, k0);
    v16h bn2 = loadB(2, k0), bn3 = loadB(3, k0);
    acc[0] = __builtin_amdgcn_wmma_f32_16x16x32_f16(false, a, false, b0,
                                                    (short)0, acc[0], false, false);
    acc[1] = __builtin_amdgcn_wmma_f32_16x16x32_f16(false, a, false, b1,
                                                    (short)0, acc[1], false, false);
    acc[2] = __builtin_amdgcn_wmma_f32_16x16x32_f16(false, a, false, b2,
                                                    (short)0, acc[2], false, false);
    acc[3] = __builtin_amdgcn_wmma_f32_16x16x32_f16(false, a, false, b3,
                                                    (short)0, acc[3], false, false);
    a = an; b0 = bn0; b1 = bn1; b2 = bn2; b3 = bn3;
  }
  // epilogue chunk
  acc[0] = __builtin_amdgcn_wmma_f32_16x16x32_f16(false, a, false, b0,
                                                  (short)0, acc[0], false, false);
  acc[1] = __builtin_amdgcn_wmma_f32_16x16x32_f16(false, a, false, b1,
                                                  (short)0, acc[1], false, false);
  acc[2] = __builtin_amdgcn_wmma_f32_16x16x32_f16(false, a, false, b2,
                                                  (short)0, acc[2], false, false);
  acc[3] = __builtin_amdgcn_wmma_f32_16x16x32_f16(false, a, false, b3,
                                                  (short)0, acc[3], false, false);

  // store: C/D layout -> lanes 0-15: VGPR r = row r; lanes 16-31: row r+8
  const int rOff = hiH ? 8 : 0;
  #pragma unroll
  for (int j = 0; j < 4; ++j) {
    const int col = colBase + (j << 4) + l15;
    const float bv = bias ? bias[col] : 0.0f;
    #pragma unroll
    for (int r = 0; r < 8; ++r) {
      const int row = (tm << 4) + rOff + r;
      float v = acc[j][r] + bv;
      if (resid) v += resid[(size_t)row * N + col];
      if (outF) outF[(size_t)row * N + col] = v;
      else      outH[(size_t)row * N + col] = (_Float16)v;
    }
  }
}

// ---------------------------------------------------------------------------
// softmax over groups of 4 contiguous floats (in place)
// ---------------------------------------------------------------------------
__global__ void k_softmax4(float* __restrict__ aw, int nGroups) {
  int g = blockIdx.x * blockDim.x + threadIdx.x;
  if (g >= nGroups) return;
  float* p = aw + (size_t)g * 4;
  float a0 = p[0], a1 = p[1], a2 = p[2], a3 = p[3];
  float m = fmaxf(fmaxf(a0, a1), fmaxf(a2, a3));
  float e0 = __expf(a0 - m), e1 = __expf(a1 - m);
  float e2 = __expf(a2 - m), e3 = __expf(a3 - m);
  float s = 1.0f / (e0 + e1 + e2 + e3);
  p[0] = e0 * s; p[1] = e1 * s; p[2] = e2 * s; p[3] = e3 * s;
}

// ---------------------------------------------------------------------------
// MSDA: one wave per (q, head); lane = channel d (0..31).
// v: [NB, NQ, HH, DH] f32, so: [NQ,128] f32 (bias added), aw: [NQ,64] softmaxed
// rp: [NQ,2]. Output: f16 [NQ, 256] = mean over NB of weighted bilinear sums.
// ---------------------------------------------------------------------------
__global__ void __launch_bounds__(256)
k_msda(const float* __restrict__ v,
       const float* __restrict__ so,
       const float* __restrict__ aw,
       const float* __restrict__ rp,
       _Float16* __restrict__ out) {
  const int lane = threadIdx.x & 31;
  const int wave = (int)((blockIdx.x * blockDim.x + threadIdx.x) >> 5);
  if (wave >= NQ * HH) return;
  const int q = wave >> 3;
  const int h = wave & 7;

  const float rx = rp[q * 2 + 0] * (float)GW;   // pre-scale by grid dims
  const float ry = rp[q * 2 + 1] * (float)GH;

  float accv = 0.0f;
  #pragma unroll
  for (int b = 0; b < NB; ++b) {
    const float* vb = v + (size_t)b * NQ * EMB;
    #pragma unroll
    for (int p = 0; p < PP; ++p) {
      const int cso = h * 16 + b * 8 + p * 2;
      const float sx = so[(size_t)q * 128 + cso + 0];
      const float sy = so[(size_t)q * 128 + cso + 1];
      const float w  = aw[(size_t)q * 64 + h * 8 + b * 4 + p];
      // px = loc.x * W - 0.5 = rx + sx - 0.5  (offset_normalizer == [W,H])
      const float px = rx + sx - 0.5f;
      const float py = ry + sy - 0.5f;
      const float x0f = floorf(px), y0f = floorf(py);
      const int x0 = (int)x0f, y0 = (int)y0f;
      const float fx = px - x0f, fy = py - y0f;

      float c00 = 0.0f, c10 = 0.0f, c01 = 0.0f, c11 = 0.0f;
      const bool xin0 = (x0 >= 0) & (x0 < GW);
      const bool xin1 = (x0 + 1 >= 0) & (x0 + 1 < GW);
      const bool yin0 = (y0 >= 0) & (y0 < GH);
      const bool yin1 = (y0 + 1 >= 0) & (y0 + 1 < GH);
      if (xin0 & yin0)
        c00 = vb[((size_t)(y0 * GW + x0) * HH + h) * DH + lane];
      if (xin1 & yin0)
        c10 = vb[((size_t)(y0 * GW + x0 + 1) * HH + h) * DH + lane];
      if (xin0 & yin1)
        c01 = vb[((size_t)((y0 + 1) * GW + x0) * HH + h) * DH + lane];
      if (xin1 & yin1)
        c11 = vb[((size_t)((y0 + 1) * GW + x0 + 1) * HH + h) * DH + lane];

      const float s = (1.0f - fx) * (1.0f - fy) * c00 + fx * (1.0f - fy) * c10
                    + (1.0f - fx) * fy * c01 + fx * fy * c11;
      accv += w * s;
    }
  }
  out[(size_t)q * EMB + h * DH + lane] = (_Float16)(0.5f * accv);  // mean(NB=2)
}

// ---------------------------------------------------------------------------
// host-side launch
// ---------------------------------------------------------------------------
extern "C" void kernel_launch(void* const* d_in, const int* in_sizes, int n_in,
                              void* d_out, int out_size, void* d_ws, size_t ws_size,
                              hipStream_t stream) {
  (void)in_sizes; (void)n_in; (void)out_size; (void)ws_size;

  const float* query   = (const float*)d_in[0];   // [NQ, 256]
  const float* value   = (const float*)d_in[1];   // [2*NQ, 256]
  const float* rp      = (const float*)d_in[2];   // [NQ, 1, 2]
  const float* W_vproj = (const float*)d_in[3];   // [256, 256]
  const float* b_vproj = (const float*)d_in[4];   // [256]
  const float* W_so    = (const float*)d_in[5];   // [512, 128]
  const float* b_so    = (const float*)d_in[6];   // [128]
  const float* W_aw    = (const float*)d_in[7];   // [512, 64]
  const float* b_aw    = (const float*)d_in[8];   // [64]
  const float* W_out   = (const float*)d_in[9];   // [256, 256]
  const float* b_out   = (const float*)d_in[10];  // [256]
  float* out = (float*)d_out;                     // [NQ, 256]

  // bump-allocate workspace (256B aligned)
  char* wp = (char*)d_ws;
  auto alloc = [&](size_t bytes) -> char* {
    char* p = wp;
    wp += (bytes + 255) & ~(size_t)255;
    return p;
  };
  _Float16* value16 = (_Float16*)alloc((size_t)2 * NQ * EMB * 2);
  _Float16* qe16    = (_Float16*)alloc((size_t)NQ * QE_K * 2);
  _Float16* WtV     = (_Float16*)alloc((size_t)EMB * EMB * 2);
  _Float16* WtSo    = (_Float16*)alloc((size_t)128 * QE_K * 2);
  _Float16* WtAw    = (_Float16*)alloc((size_t)64 * QE_K * 2);
  _Float16* WtOut   = (_Float16*)alloc((size_t)EMB * EMB * 2);
  float*    vproj   = (float*)alloc((size_t)2 * NQ * EMB * 4);
  float*    soBuf   = (float*)alloc((size_t)NQ * 128 * 4);
  float*    awBuf   = (float*)alloc((size_t)NQ * 64 * 4);
  _Float16* msda16  = (_Float16*)alloc((size_t)NQ * EMB * 2);

  const int TB = 256;
  // --- prep: conversions / concat / weight transposes ---
  {
    int n = 2 * NQ * EMB;
    k_f32_to_f16<<<(n + TB - 1) / TB, TB, 0, stream>>>(value, value16, n);
  }
  {
    int n = NQ * QE_K;
    k_build_qe<<<(n + TB - 1) / TB, TB, 0, stream>>>(value, query, qe16);
  }
  k_transpose_f16<<<(EMB * EMB + TB - 1) / TB, TB, 0, stream>>>(W_vproj, WtV, EMB, EMB);
  k_transpose_f16<<<(128 * QE_K + TB - 1) / TB, TB, 0, stream>>>(W_so, WtSo, QE_K, 128);
  k_transpose_f16<<<(64 * QE_K + TB - 1) / TB, TB, 0, stream>>>(W_aw, WtAw, QE_K, 64);
  k_transpose_f16<<<(EMB * EMB + TB - 1) / TB, TB, 0, stream>>>(W_out, WtOut, EMB, EMB);

  // --- GEMM 1: v = value @ W_vproj + b_vproj  (f32 out) ---
  {
    int M = 2 * NQ, N = EMB, K = EMB;
    int waves = (M >> 4) * (N >> 6);
    k_gemm_wmma<<<(waves * 32 + TB - 1) / TB, TB, 0, stream>>>(
        value16, WtV, b_vproj, nullptr, vproj, nullptr, M, N, K);
  }
  // --- GEMM 2: so = qe @ W_so + b_so ---
  {
    int M = NQ, N = 128, K = QE_K;
    int waves = (M >> 4) * (N >> 6);
    k_gemm_wmma<<<(waves * 32 + TB - 1) / TB, TB, 0, stream>>>(
        qe16, WtSo, b_so, nullptr, soBuf, nullptr, M, N, K);
  }
  // --- GEMM 3: aw_logits = qe @ W_aw + b_aw ---
  {
    int M = NQ, N = 64, K = QE_K;
    int waves = (M >> 4) * (N >> 6);
    k_gemm_wmma<<<(waves * 32 + TB - 1) / TB, TB, 0, stream>>>(
        qe16, WtAw, b_aw, nullptr, awBuf, nullptr, M, N, K);
  }
  // --- softmax over L*P = 4 ---
  {
    int groups = NQ * HH * NB;
    k_softmax4<<<(groups + TB - 1) / TB, TB, 0, stream>>>(awBuf, groups);
  }
  // --- MSDA (bilinear gather + weighted sum, mean over NB) ---
  {
    int waves = NQ * HH;
    k_msda<<<(waves * 32 + TB - 1) / TB, TB, 0, stream>>>(
        vproj, soBuf, awBuf, rp, msda16);
  }
  // --- GEMM 4: out = msda @ W_out + b_out + query ---
  {
    int M = NQ, N = EMB, K = EMB;
    int waves = (M >> 4) * (N >> 6);
    k_gemm_wmma<<<(waves * 32 + TB - 1) / TB, TB, 0, stream>>>(
        msda16, WtOut, b_out, query, out, nullptr, M, N, K);
  }
}
